// TripletBlock_63539746177154
// MI455X (gfx1250) — matrix operand order repaired and checked by reading, not compiled, not run on hardware
//
#include <hip/hip_runtime.h>
#include <hip/hip_bf16.h>

typedef __bf16 bf16;
typedef __attribute__((ext_vector_type(16))) __bf16 v16bf;
typedef __attribute__((ext_vector_type(8)))  float  v8f;

#define DEV static __device__ __forceinline__

DEV v8f wmma_bf16(v16bf a, v16bf b, v8f c) {
  // D = A(16x32) * B(32x16) + C, f32 accumulate
  return __builtin_amdgcn_wmma_f32_16x16x32_bf16(false, a, false, b, (short)0, c,
                                                 false, false);
}

DEV v16bf load_fragA(const bf16* p) {
  // A layout: halves 0..15 = 16 consecutive K starting at hi*16
  union { v16bf v; uint4 q[2]; } u;
  u.q[0] = *(const uint4*)p;
  u.q[1] = *(const uint4*)(p + 8);
  return u.v;
}
DEV v16bf load_fragB(const bf16* p) {
  // B layout: halves 0..15 = 16 consecutive K starting at hi*16
  union { v16bf v; uint4 q[2]; } u;
  u.q[0] = *(const uint4*)p;
  u.q[1] = *(const uint4*)(p + 8);
  return u.v;
}
DEV v8f vzero() { v8f z = {0.f,0.f,0.f,0.f,0.f,0.f,0.f,0.f}; return z; }

// CDNA5 async DMA: global -> LDS, tracked by ASYNCcnt
DEV void async_copy16(unsigned lds_byte_off, const bf16* g) {
  asm volatile("global_load_async_to_lds_b128 %0, %1, off"
               :: "v"(lds_byte_off), "v"(g) : "memory");
}
#define WAIT_ASYNC0() asm volatile("s_wait_asynccnt 0" ::: "memory")
#define WAIT_ASYNC6() asm volatile("s_wait_asynccnt 6" ::: "memory")

union H8 { uint4 q; bf16 h[8]; };

// ---------------------------------------------------------------- LayerNorm
__global__ __launch_bounds__(256) void ln_kernel(const float* __restrict__ x,
                                                 const float* __restrict__ w,
                                                 const float* __restrict__ b,
                                                 bf16* __restrict__ y) {
  int wave = threadIdx.x >> 5, lane = threadIdx.x & 31;
  size_t tok = (size_t)blockIdx.x * 8 + wave;
  const float* xp = x + tok * 256 + lane * 8;
  float4 a = *(const float4*)xp;
  float4 c = *(const float4*)(xp + 4);
  float v[8] = {a.x, a.y, a.z, a.w, c.x, c.y, c.z, c.w};
  float s = 0.f, sq = 0.f;
#pragma unroll
  for (int i = 0; i < 8; ++i) { s += v[i]; sq += v[i] * v[i]; }
#pragma unroll
  for (int m = 16; m >= 1; m >>= 1) {
    s  += __shfl_xor(s, m, 32);
    sq += __shfl_xor(sq, m, 32);
  }
  float mean = s * (1.f / 256.f);
  float var  = sq * (1.f / 256.f) - mean * mean;
  float rstd = rsqrtf(var + 1e-5f);
  H8 o0, o1;
#pragma unroll
  for (int i = 0; i < 4; ++i) {
    int ch = lane * 8 + i;
    o0.h[i] = (bf16)((v[i] - mean) * rstd * w[ch] + b[ch]);
  }
#pragma unroll
  for (int i = 4; i < 8; ++i) {
    int ch = lane * 8 + i;
    o0.h[i] = (bf16)((v[i] - mean) * rstd * w[ch] + b[ch]);
  }
  (void)o1;
  *(uint4*)(y + tok * 256 + lane * 8) = o0.q;
}

// ------------------------------------------------------------- f32 -> bf16
__global__ __launch_bounds__(256) void cvt_kernel(const float* __restrict__ in,
                                                  bf16* __restrict__ out, int n) {
  int i = blockIdx.x * 256 + threadIdx.x;
  if (i < n) out[i] = (bf16)in[i];
}

// ------------------------------------------- relative position bias (tiny MLP)
__global__ __launch_bounds__(256) void rp_table_kernel(const float* __restrict__ w1,
                                                       const float* __restrict__ b1,
                                                       const float* __restrict__ w2,
                                                       const float* __restrict__ b2,
                                                       float* __restrict__ table) {
  int gid = blockIdx.x * 256 + threadIdx.x;
  if (gid >= 225 * 8) return;
  int row = gid >> 3, hh = gid & 7;
  float cy = (float)(row / 15 - 7), cx = (float)(row % 15 - 7);
  float t0 = (cy > 0.f) ? log1pf(cy) : (cy < 0.f ? -log1pf(-cy) : 0.f);
  float t1 = (cx > 0.f) ? log1pf(cx) : (cx < 0.f ? -log1pf(-cx) : 0.f);
  float acc = b2[hh];
  for (int h = 0; h < 512; ++h) {
    float hv = fmaf(t0, w1[h * 2], fmaf(t1, w1[h * 2 + 1], b1[h]));
    hv = fmaxf(hv, 0.f);
    acc = fmaf(hv, w2[hh * 512 + h], acc);
  }
  table[row * 8 + hh] = acc;
}

// Writes bias swizzled for the attention kernel's float4 loads:
// bias_sw[ ((h*64 + i)*16 + (j&15)) * 4 + (j>>4) ] = bias[h][i][j]
__global__ __launch_bounds__(256) void rp_expand_kernel(const float* __restrict__ table,
                                                        float* __restrict__ bias) {
  int gid = blockIdx.x * 256 + threadIdx.x;  // [hh][i][j], 8*64*64
  int j = gid & 63, i = (gid >> 6) & 63, hh = gid >> 12;
  int idx = ((i >> 3) - (j >> 3) + 7) * 15 + ((i & 7) - (j & 7) + 7);
  bias[((((hh << 6) + i) * 16 + (j & 15)) << 2) + (j >> 4)] = table[idx * 8 + hh];
}

// --------------------------------------------------------------- bf16 GEMM
// out[M,N] = A[M,K] * W[N,K]^T (+bias). Block tile 128x256, 8 waves (2x4),
// K pipelined 32-wide through double-buffered LDS via async DMA.
// MODE 0: bf16 out, no bias (QKV)
// MODE 1: X[m,n] += ls[n]*(d+bias[n]) (proj/fc2 residual, f32)
// MODE 2: bf16 out = GELU(d+bias) (fc1)
template <int MODE>
__global__ __launch_bounds__(256) void gemm_bf16_kernel(
    const bf16* __restrict__ A, const bf16* __restrict__ W,
    const float* __restrict__ bias, const float* __restrict__ ls,
    float* __restrict__ X, bf16* __restrict__ out, int M, int N, int K) {
  __shared__ __align__(16) bf16 Abuf[2][128 * 32];   // 2 x 8KB
  __shared__ __align__(16) bf16 Bbuf[2][256 * 32];   // 2 x 16KB
  int tid = threadIdx.x;
  int wave = tid >> 5, lane = tid & 31;
  int wm = wave >> 2, wn = wave & 3;                 // 2 x 4 wave grid
  int nBlkN = N >> 8;
  int bm = blockIdx.x / nBlkN, bn = blockIdx.x - bm * nBlkN;
  int m0b = bm << 7, n0b = bn << 8;
  const bf16* Ab = A + (size_t)m0b * K;
  const bf16* Wb = W + (size_t)n0b * K;
  int c = lane & 15, hi = lane >> 4;

  auto stage = [&](int bufIdx, int k) {
    unsigned la = (unsigned)(uintptr_t)&Abuf[bufIdx][0];
    unsigned lb = (unsigned)(uintptr_t)&Bbuf[bufIdx][0];
#pragma unroll
    for (int u = 0; u < 2; ++u) {                    // A: 512 16B chunks
      int ch = tid * 2 + u;
      int row = ch >> 2, q = ch & 3;
      async_copy16(la + (unsigned)(row * 64 + q * 16),
                   Ab + (size_t)row * K + k + q * 8);
    }
#pragma unroll
    for (int u = 0; u < 4; ++u) {                    // B: 1024 16B chunks
      int ch = tid * 4 + u;
      int row = ch >> 2, q = ch & 3;
      async_copy16(lb + (unsigned)(row * 64 + q * 16),
                   Wb + (size_t)row * K + k + q * 8);
    }
  };

  v8f acc[4][4];
#pragma unroll
  for (int i = 0; i < 4; ++i)
#pragma unroll
    for (int j = 0; j < 4; ++j) acc[i][j] = vzero();

  int nk = K >> 5;
  stage(0, 0);
  for (int kt = 0; kt < nk; ++kt) {
    int cur = kt & 1;
    if (kt + 1 < nk) {
      stage(1 - cur, (kt + 1) << 5);   // overlap next DMA with this compute
      WAIT_ASYNC6();                    // oldest 6 (current buffer) done
    } else {
      WAIT_ASYNC0();
    }
    __syncthreads();
    const bf16* As = &Abuf[cur][0];
    const bf16* Bs = &Bbuf[cur][0];
    v16bf af[4];
#pragma unroll
    for (int i = 0; i < 4; ++i)
      af[i] = load_fragA(As + (wm * 64 + i * 16 + c) * 32 + (hi << 4));
#pragma unroll
    for (int j = 0; j < 4; ++j) {
      v16bf bw = load_fragB(Bs + (wn * 64 + j * 16 + c) * 32 + (hi << 4));
#pragma unroll
      for (int i = 0; i < 4; ++i) acc[i][j] = wmma_bf16(af[i], bw, acc[i][j]);
    }
    __syncthreads();                    // buffer free before next overwrite
  }

  int m0 = m0b + wm * 64, n0 = n0b + wn * 64;
#pragma unroll
  for (int j = 0; j < 4; ++j) {
    int n = n0 + j * 16 + c;
    float bj = (MODE == 0) ? 0.f : bias[n];
    float lj = (MODE == 1) ? ls[n] : 0.f;
#pragma unroll
    for (int i = 0; i < 4; ++i) {
      int mbase = m0 + i * 16 + (hi << 3);
#pragma unroll
      for (int e = 0; e < 8; ++e) {
        size_t idx = (size_t)(mbase + e) * N + n;
        float d = acc[i][j][e] + bj;
        if (MODE == 1) {
          X[idx] = X[idx] + lj * d;
        } else if (MODE == 2) {
          out[idx] = (bf16)(0.5f * d * (1.f + erff(d * 0.70710678118654752f)));
        } else {
          out[idx] = (bf16)d;
        }
      }
    }
  }
}

// ------------------------------------------------------ windowed attention
DEV int win_token(int wi, int l, int mode) {
  int bt = wi >> 6, hb = (wi >> 3) & 7, wb = wi & 7;
  int gy = l >> 3, gx = l & 7;
  int h, w;
  if (mode == 0) { h = hb * 8 + gy; w = wb * 8 + gx; }   // window partition
  else           { h = gy * 8 + hb; w = gx * 8 + wb; }   // grid partition
  return bt * 4096 + h * 64 + w;
}

__global__ __launch_bounds__(256) void win_attn_kernel(const bf16* __restrict__ qkv,
                                                       const float* __restrict__ bias,
                                                       bf16* __restrict__ out,
                                                       int mode) {
  __shared__ __align__(16) bf16 Pbuf[8][16 * 64];
  __shared__ __align__(16) bf16 Vbuf[8][64 * 32];
  int hh = threadIdx.x >> 5, lane = threadIdx.x & 31;
  int wi = blockIdx.x;
  int c = lane & 15, hi = lane >> 4, hi8 = hi << 3;
  bf16* Pw = &Pbuf[hh][0];
  bf16* Vw = &Vbuf[hh][0];

  // K^T fragments (B operand), kept live across row tiles
  v16bf bk[4];
#pragma unroll
  for (int i = 0; i < 4; ++i) {
    int tok = win_token(wi, i * 16 + c, mode);
    bk[i] = load_fragB(qkv + (size_t)tok * 768 + 256 + hh * 32 + (hi << 4));
  }
  // Stage V (64x32) into LDS, then build B fragments of V
#pragma unroll
  for (int rr = 0; rr < 2; ++rr) {
    int l = lane * 2 + rr;
    int tok = win_token(wi, l, mode);
    const uint4* src = (const uint4*)(qkv + (size_t)tok * 768 + 512 + hh * 32);
    uint4* dst = (uint4*)(Vw + l * 32);
    dst[0] = src[0]; dst[1] = src[1]; dst[2] = src[2]; dst[3] = src[3];
  }
  v16bf bv[2][2];
#pragma unroll
  for (int nv = 0; nv < 2; ++nv)
#pragma unroll
    for (int kc = 0; kc < 2; ++kc) {
      v16bf t;
#pragma unroll
      for (int kk = 0; kk < 16; ++kk)
        t[kk] = Vw[(kc * 32 + (hi << 4) + kk) * 32 + nv * 16 + c];
      bv[nv][kc] = t;
    }

  const float scale = 0.17677669529663687f;  // 1/sqrt(32)
  const float* bh = bias + (hh << 12);       // swizzled [m][c][nj]

  // Stream over 16-row tiles: only one S strip live at a time (no spills)
  for (int mi = 0; mi < 4; ++mi) {
    int tokm = win_token(wi, mi * 16 + c, mode);
    v16bf aq = load_fragA(qkv + (size_t)tokm * 768 + hh * 32 + (hi << 4));
    v8f s[4];
#pragma unroll
    for (int nj = 0; nj < 4; ++nj) s[nj] = vzero();
#pragma unroll
    for (int nj = 0; nj < 4; ++nj) s[nj] = wmma_bf16(aq, bk[nj], s[nj]);

#pragma unroll
    for (int e = 0; e < 8; ++e) {
      int m = mi * 16 + e + hi8;
      float4 bq = *(const float4*)(bh + ((m * 16 + c) << 2));  // nj=0..3
      float p0 = fmaf(s[0][e], scale, bq.x);
      float p1 = fmaf(s[1][e], scale, bq.y);
      float p2 = fmaf(s[2][e], scale, bq.z);
      float p3 = fmaf(s[3][e], scale, bq.w);
      float mx = fmaxf(fmaxf(p0, p1), fmaxf(p2, p3));
#pragma unroll
      for (int msk = 8; msk >= 1; msk >>= 1) mx = fmaxf(mx, __shfl_xor(mx, msk, 32));
      p0 = __expf(p0 - mx); p1 = __expf(p1 - mx);
      p2 = __expf(p2 - mx); p3 = __expf(p3 - mx);
      float sum = (p0 + p1) + (p2 + p3);
#pragma unroll
      for (int msk = 8; msk >= 1; msk >>= 1) sum += __shfl_xor(sum, msk, 32);
      float inv = 1.f / sum;
      bf16* pr = Pw + (e + hi8) * 64 + c;
      pr[0]  = (bf16)(p0 * inv);
      pr[16] = (bf16)(p1 * inv);
      pr[32] = (bf16)(p2 * inv);
      pr[48] = (bf16)(p3 * inv);
    }
    // P row-tile (16x64) back as A fragments, O = P * V
    v16bf ap[2];
#pragma unroll
    for (int kc = 0; kc < 2; ++kc)
      ap[kc] = load_fragA(Pw + c * 64 + kc * 32 + (hi << 4));
#pragma unroll
    for (int nv = 0; nv < 2; ++nv) {
      v8f o = vzero();
      o = wmma_bf16(ap[0], bv[nv][0], o);
      o = wmma_bf16(ap[1], bv[nv][1], o);
#pragma unroll
      for (int e = 0; e < 8; ++e) {
        int m = mi * 16 + e + hi8;
        int tok = win_token(wi, m, mode);
        out[(size_t)tok * 256 + hh * 32 + nv * 16 + c] = (bf16)o[e];
      }
    }
  }
}

// ------------------------------------------------- causal time attention (t=8)
DEV void load32f(const bf16* p, float* dst) {
#pragma unroll
  for (int c4 = 0; c4 < 4; ++c4) {
    H8 u; u.q = *(const uint4*)(p + c4 * 8);
#pragma unroll
    for (int d = 0; d < 8; ++d) dst[c4 * 8 + d] = (float)u.h[d];
  }
}

__global__ __launch_bounds__(256) void time_attn_kernel(const bf16* __restrict__ qkv,
                                                        bf16* __restrict__ out) {
  int gid = blockIdx.x * 256 + threadIdx.x;  // 2*4096*8heads*8t = 524288
  int tt = gid & 7, hh = (gid >> 3) & 7, sn = gid >> 6;
  int b = sn >> 12, n = sn & 4095;
  size_t tokQ = (size_t)(b * 8 + tt) * 4096 + n;
  float q[32];
  load32f(qkv + tokQ * 768 + hh * 32, q);
  const float scale = 0.17677669529663687f;
  float sc[8];
  float mx = -3.0e38f;
  for (int j = 0; j <= tt; ++j) {
    float kv[32];
    load32f(qkv + ((size_t)(b * 8 + j) * 4096 + n) * 768 + 256 + hh * 32, kv);
    float s = 0.f;
#pragma unroll
    for (int d = 0; d < 32; ++d) s = fmaf(q[d], kv[d], s);
    sc[j] = s * scale;
    mx = fmaxf(mx, sc[j]);
  }
  float sum = 0.f;
  for (int j = 0; j <= tt; ++j) { sc[j] = __expf(sc[j] - mx); sum += sc[j]; }
  float inv = 1.f / sum;
  float o[32];
#pragma unroll
  for (int d = 0; d < 32; ++d) o[d] = 0.f;
  for (int j = 0; j <= tt; ++j) {
    float vv[32];
    load32f(qkv + ((size_t)(b * 8 + j) * 4096 + n) * 768 + 512 + hh * 32, vv);
    float p = sc[j] * inv;
#pragma unroll
    for (int d = 0; d < 32; ++d) o[d] = fmaf(p, vv[d], o[d]);
  }
  bf16* op = out + tokQ * 256 + hh * 32;
#pragma unroll
  for (int c4 = 0; c4 < 4; ++c4) {
    H8 u;
#pragma unroll
    for (int d = 0; d < 8; ++d) u.h[d] = (bf16)o[c4 * 8 + d];
    *(uint4*)(op + c4 * 8) = u.q;
  }
}

// ------------------------------------------------------------------- host
static constexpr size_t OFF_Y     = 0;                         // 32MB bf16 LN out
static constexpr size_t OFF_QKV   = 33554432;                  // 96MB bf16 qkv
static constexpr size_t OFF_ATTN  = 134217728;                 // 32MB bf16 attn out
static constexpr size_t OFF_H     = 33554432;                  // 128MB, aliases qkv+attn (safe)
static constexpr size_t OFF_WQKV  = 167772160;
static constexpr size_t OFF_WPROJ = OFF_WQKV + (size_t)768 * 256 * 2;
static constexpr size_t OFF_WFC1  = OFF_WPROJ + (size_t)256 * 256 * 2;
static constexpr size_t OFF_WFC2  = OFF_WFC1 + (size_t)1024 * 256 * 2;
static constexpr size_t OFF_TAB   = OFF_WFC2 + (size_t)256 * 1024 * 2;
static constexpr size_t OFF_BIAS  = OFF_TAB + 8192;

static void run_spatial(void* const* P, float* x, char* ws, int mode, hipStream_t st) {
  auto f = [&](int i) { return (const float*)P[i]; };
  bf16* y     = (bf16*)(ws + OFF_Y);
  bf16* qkv   = (bf16*)(ws + OFF_QKV);
  bf16* attn  = (bf16*)(ws + OFF_ATTN);
  bf16* h     = (bf16*)(ws + OFF_H);
  bf16* wqkv  = (bf16*)(ws + OFF_WQKV);
  bf16* wproj = (bf16*)(ws + OFF_WPROJ);
  bf16* wfc1  = (bf16*)(ws + OFF_WFC1);
  bf16* wfc2  = (bf16*)(ws + OFF_WFC2);
  float* tab  = (float*)(ws + OFF_TAB);
  float* bias = (float*)(ws + OFF_BIAS);

  rp_table_kernel<<<8, 256, 0, st>>>(f(5), f(6), f(7), f(8), tab);
  rp_expand_kernel<<<128, 256, 0, st>>>(tab, bias);
  cvt_kernel<<<(768 * 256 + 255) / 256, 256, 0, st>>>(f(2), wqkv, 768 * 256);
  cvt_kernel<<<(256 * 256 + 255) / 256, 256, 0, st>>>(f(3), wproj, 256 * 256);
  cvt_kernel<<<(1024 * 256 + 255) / 256, 256, 0, st>>>(f(12), wfc1, 1024 * 256);
  cvt_kernel<<<(256 * 1024 + 255) / 256, 256, 0, st>>>(f(14), wfc2, 256 * 1024);

  ln_kernel<<<8192, 256, 0, st>>>(x, f(0), f(1), y);
  gemm_bf16_kernel<0><<<1536, 256, 0, st>>>(y, wqkv, nullptr, nullptr, nullptr, qkv,
                                            65536, 768, 256);
  win_attn_kernel<<<1024, 256, 0, st>>>(qkv, bias, attn, mode);
  gemm_bf16_kernel<1><<<512, 256, 0, st>>>(attn, wproj, f(4), f(9), x, nullptr,
                                           65536, 256, 256);
  ln_kernel<<<8192, 256, 0, st>>>(x, f(10), f(11), y);
  gemm_bf16_kernel<2><<<2048, 256, 0, st>>>(y, wfc1, f(13), nullptr, nullptr, h,
                                            65536, 1024, 256);
  gemm_bf16_kernel<1><<<512, 256, 0, st>>>(h, wfc2, f(15), f(16), x, nullptr,
                                           65536, 256, 1024);
}

static void run_time(void* const* P, float* x, char* ws, hipStream_t st) {
  auto f = [&](int i) { return (const float*)P[i]; };
  bf16* y     = (bf16*)(ws + OFF_Y);
  bf16* qkv   = (bf16*)(ws + OFF_QKV);
  bf16* attn  = (bf16*)(ws + OFF_ATTN);
  bf16* h     = (bf16*)(ws + OFF_H);
  bf16* wqkv  = (bf16*)(ws + OFF_WQKV);
  bf16* wproj = (bf16*)(ws + OFF_WPROJ);
  bf16* wfc1  = (bf16*)(ws + OFF_WFC1);
  bf16* wfc2  = (bf16*)(ws + OFF_WFC2);

  cvt_kernel<<<(768 * 256 + 255) / 256, 256, 0, st>>>(f(2), wqkv, 768 * 256);
  cvt_kernel<<<(256 * 256 + 255) / 256, 256, 0, st>>>(f(3), wproj, 256 * 256);
  cvt_kernel<<<(1024 * 256 + 255) / 256, 256, 0, st>>>(f(8), wfc1, 1024 * 256);
  cvt_kernel<<<(256 * 1024 + 255) / 256, 256, 0, st>>>(f(10), wfc2, 256 * 1024);

  ln_kernel<<<8192, 256, 0, st>>>(x, f(0), f(1), y);
  gemm_bf16_kernel<0><<<1536, 256, 0, st>>>(y, wqkv, nullptr, nullptr, nullptr, qkv,
                                            65536, 768, 256);
  time_attn_kernel<<<2048, 256, 0, st>>>(qkv, attn);
  gemm_bf16_kernel<1><<<512, 256, 0, st>>>(attn, wproj, f(4), f(5), x, nullptr,
                                           65536, 256, 256);
  ln_kernel<<<8192, 256, 0, st>>>(x, f(6), f(7), y);
  gemm_bf16_kernel<2><<<2048, 256, 0, st>>>(y, wfc1, f(9), nullptr, nullptr, h,
                                            65536, 1024, 256);
  gemm_bf16_kernel<1><<<512, 256, 0, st>>>(h, wfc2, f(11), f(12), x, nullptr,
                                           65536, 256, 1024);
}

extern "C" void kernel_launch(void* const* d_in, const int* in_sizes, int n_in,
                              void* d_out, int out_size, void* d_ws, size_t ws_size,
                              hipStream_t stream) {
  (void)in_sizes; (void)n_in; (void)out_size; (void)ws_size;
  float* x = (float*)d_out;           // running residual stream (f32)
  char* ws = (char*)d_ws;
  hipMemcpyAsync(x, d_in[0], (size_t)16 * 4096 * 256 * sizeof(float),
                 hipMemcpyDeviceToDevice, stream);
  // d_in layout: 0=x, 1..17 win params, 18..34 grid params, 35..47 time params, 48..50 t/H/W
  run_spatial(d_in + 1, x, ws, 0, stream);   // window partition
  run_spatial(d_in + 18, x, ws, 1, stream);  // grid partition
  run_time(d_in + 35, x, ws, stream);        // causal time block
}
